// EnergyTransformer_3410204033891
// MI455X (gfx1250) — matrix operand order repaired and checked by reading, not compiled, not run on hardware
//
#include <hip/hip_runtime.h>
#include <hip/hip_bf16.h>
#include <cstdint>

// Problem dims (fixed by reference)
#define B_  8
#define N_  196
#define D_  768
#define H_  12
#define Y_  64
#define M_  3072
#define T_  (B_*N_)                 // 1568 tokens
#define TD  ((size_t)T_*D_)         // 1,204,224
#define TLD 208                     // padded leading dim for 196-wide tensors (16B-aligned rows)
static constexpr float BETA_C   = 0.125f;   // 1/sqrt(64)
static constexpr float LN_EPS_C = 1e-5f;

typedef __attribute__((ext_vector_type(16))) __bf16 bf16x16;
typedef __attribute__((ext_vector_type(8)))  float  f32x8;
typedef __attribute__((ext_vector_type(4)))  int    v4i;

struct F4x2 { float4 lo; float4 hi; };

__device__ __forceinline__ unsigned short f2bf(float f) {
  unsigned int u = __builtin_bit_cast(unsigned int, f);
  unsigned int r = (u + 0x7FFFu + ((u >> 16) & 1u)) >> 16;   // RNE
  return (unsigned short)r;
}
__device__ __forceinline__ __bf16 bfzero() {
  return __builtin_bit_cast(__bf16, (unsigned short)0);
}

// ---------------------------------------------------------------------------
// 16B global -> LDS copy via the CDNA5 async-to-LDS path (ASYNCcnt);
// falls back to vectorized load+store if the builtin is unavailable.
// ---------------------------------------------------------------------------
#if defined(__HIP_DEVICE_COMPILE__) && __has_builtin(__builtin_amdgcn_global_load_async_to_lds_b128)
#define HAVE_ASYNC 1
typedef __attribute__((address_space(1))) v4i* gptr_v4i;
typedef __attribute__((address_space(3))) v4i* lptr_v4i;
#else
#define HAVE_ASYNC 0
#endif

__device__ __forceinline__ void cp16_g2l(const __bf16* gptr, __bf16* lptr) {
#if HAVE_ASYNC
  __builtin_amdgcn_global_load_async_to_lds_b128(
      (gptr_v4i)(unsigned long long)(uintptr_t)gptr,
      (lptr_v4i)(unsigned int)(uintptr_t)lptr, 0, 0);
#else
  *(float4*)lptr = *(const float4*)gptr;
#endif
}
// Wait until ASYNCcnt <= NCNT (async loads complete in order, so waiting for
// "<= number just issued for the NEXT tile" guarantees the current tile is in LDS).
template<int NCNT>
__device__ __forceinline__ void cp_wait_n() {
#if HAVE_ASYNC
#if __has_builtin(__builtin_amdgcn_s_wait_asynccnt)
  __builtin_amdgcn_s_wait_asynccnt(NCNT);
#else
  asm volatile("s_wait_asynccnt %0" :: "i"(NCNT) : "memory");
#endif
#endif
}

// ---------------------------------------------------------------------------
// WMMA GEMM (pure NT): C[M,N] (+)= alpha * A[M,K] * B[N,K]^T
//   A, B bf16 row-major, K-contiguous. Double-buffered LDS; interior tiles are
//   staged with async b128 copies overlapped with the previous tile's WMMAs.
//   ACC: accumulate into existing f32 C (MODE 0 only)
//   MODE: 0 = store f32
//         1 = store bf16
//         2 = no store; atomicAdd(e_slot, -0.5*sum relu(v)^2)
//         3 = store relu(v) bf16 AND the MODE-2 energy accumulation
//         4 = store bf16 to C and transposed-per-batch bf16 to C2
//             (C2[b][n][m%196] with leading dim TLD; batch = m/196)
// Batched via blockIdx.z = z1*z2n + z2 with per-operand (s*1,s*2) strides.
// ---------------------------------------------------------------------------
#define BM 64
#define BN 64
#define BK 32
#define LDSP 40

template<bool ACC, int MODE>
__global__ __launch_bounds__(128)
void gemm_nt(const __bf16* __restrict__ A, const __bf16* __restrict__ B,
             void* __restrict__ Cvoid, unsigned short* __restrict__ C2,
             int M, int N, int K, int lda, int ldb, int ldc,
             long sA1, long sA2, long sB1, long sB2, long sC1, long sC2,
             int z2n, float alpha, float* __restrict__ e_slot)
{
  __shared__ __align__(16) __bf16 As[2][BM][LDSP];
  __shared__ __align__(16) __bf16 Bs[2][BN][LDSP];

  const int tid   = threadIdx.x;
  const int wave  = tid >> 5;
  const int lane  = tid & 31;
  const int lrow  = lane & 15;
  const int lhalf = lane >> 4;
  const int wr = (wave >> 1) * 32;
  const int wc = (wave & 1)  * 32;

  const int m0 = blockIdx.x * BM;
  const int n0 = blockIdx.y * BN;
  const int z  = blockIdx.z;
  const int z1 = z / z2n, z2 = z % z2n;
  A += (size_t)z1 * sA1 + (size_t)z2 * sA2;
  B += (size_t)z1 * sB1 + (size_t)z2 * sB2;
  const size_t coff = (size_t)z1 * sC1 + (size_t)z2 * sC2;

  // Per-thread fixed 16B-chunk coordinates for staging (2 chunks per operand).
  const int rch = tid >> 2;           // chunk row 0..31
  const int kch = (tid & 3) * 8;      // chunk col offset (elements)
  const __bf16* pA0 = A + (size_t)(m0 + rch) * lda + kch;
  const __bf16* pA1 = pA0 + (size_t)32 * lda;
  const __bf16* pB0 = B + (size_t)(n0 + rch) * ldb + kch;
  const __bf16* pB1 = pB0 + (size_t)32 * ldb;
  const bool fullM = (m0 + BM <= M);
  const bool fullN = (n0 + BN <= N);

  auto stageA = [&](int kk, int b) -> int {
    if (fullM && kk + BK <= K) {
      cp16_g2l(pA0 + kk, &As[b][rch][kch]);
      cp16_g2l(pA1 + kk, &As[b][rch + 32][kch]);
      return 2;
    }
    for (int i = tid; i < BM * BK; i += 128) {
      int r = i / BK, c = i % BK;
      int gm = m0 + r, gk = kk + c;
      As[b][r][c] = (gm < M && gk < K) ? A[(size_t)gm * lda + gk] : bfzero();
    }
    return 0;
  };
  auto stageB = [&](int kk, int b) -> int {
    if (fullN && kk + BK <= K) {
      cp16_g2l(pB0 + kk, &Bs[b][rch][kch]);
      cp16_g2l(pB1 + kk, &Bs[b][rch + 32][kch]);
      return 2;
    }
    for (int i = tid; i < BN * BK; i += 128) {
      int r = i / BK, c = i % BK;
      int gn = n0 + r, gk = kk + c;
      Bs[b][r][c] = (gn < N && gk < K) ? B[(size_t)gn * ldb + gk] : bfzero();
    }
    return 0;
  };

  f32x8 acc[2][2] = {};
  const int nsteps = (K + BK - 1) / BK;

  // Prologue: stage tile 0 into buffer 0.
  stageA(0, 0);
  stageB(0, 0);

  for (int s = 0; s < nsteps; ++s) {
    const int kk  = s * BK;
    const int cur = s & 1;
    int nx = 0;
    if (s + 1 < nsteps) {
      // Issue async copies for the next tile into the other buffer; the
      // buffer was last read at step s-1 and freed by that step's barrier.
      nx = stageA(kk + BK, cur ^ 1) + stageB(kk + BK, cur ^ 1);
      if (s + 2 < nsteps) {               // L2 prefetch two tiles ahead
        __builtin_prefetch(pA0 + kk + 2 * BK, 0, 0);
        __builtin_prefetch(pB0 + kk + 2 * BK, 0, 0);
      }
    }
    // Wait for the CURRENT tile (in-order completion => <= nx outstanding).
    if (nx == 4)      cp_wait_n<4>();
    else if (nx == 2) cp_wait_n<2>();
    else              cp_wait_n<0>();
    __syncthreads();

    // ---- fragments ----
    // A frag (16x32): lane<16 -> row lrow, K {0..7,16..23}; lane>=16 -> K {8..15,24..31}
    const int kofA = lhalf * 8;
    bf16x16 afrag[2], bfrag[2];
    #pragma unroll
    for (int i = 0; i < 2; ++i) {
      const float4* p = reinterpret_cast<const float4*>(&As[cur][wr + i * 16 + lrow][kofA]);
      F4x2 f; f.lo = p[0]; f.hi = p[2];
      afrag[i] = __builtin_bit_cast(bf16x16, f);
    }
    // B frag (32x16): lane<16 -> col lrow, K {0..7,16..23}; lane>=16 -> K {8..15,24..31}
    const int kofB = lhalf * 8;
    #pragma unroll
    for (int j = 0; j < 2; ++j) {
      const float4* p = reinterpret_cast<const float4*>(&Bs[cur][wc + j * 16 + lrow][kofB]);
      F4x2 f; f.lo = p[0]; f.hi = p[2];
      bfrag[j] = __builtin_bit_cast(bf16x16, f);
    }
    #pragma unroll
    for (int i = 0; i < 2; ++i)
      #pragma unroll
      for (int j = 0; j < 2; ++j)
        acc[i][j] = __builtin_amdgcn_wmma_f32_16x16x32_bf16(
            false, afrag[i], false, bfrag[j], (short)0, acc[i][j], false, false);
    __syncthreads();   // frees buffer 'cur' for re-staging at step s+1
  }

  // ---- epilogue ----
  float* Cf          = (float*)Cvoid + coff;
  unsigned short* Cu = (unsigned short*)Cvoid + coff;
  float lsum = 0.f;
  #pragma unroll
  for (int i = 0; i < 2; ++i) {
    int mb = m0 + wr + i * 16 + lhalf * 8;
    #pragma unroll
    for (int j = 0; j < 2; ++j) {
      int n = n0 + wc + j * 16 + lrow;
      if (n < N) {
        #pragma unroll
        for (int r = 0; r < 8; ++r) {
          int m = mb + r;
          if (m < M) {
            float v = alpha * acc[i][j][r];
            size_t idx = (size_t)m * ldc + n;
            if (MODE == 0) {
              if (ACC) v += Cf[idx];
              Cf[idx] = v;
            } else if (MODE == 1) {
              Cu[idx] = f2bf(v);
            } else if (MODE == 2) {
              float rv = v > 0.f ? v : 0.f;
              lsum += rv * rv;
            } else if (MODE == 3) {
              float rv = v > 0.f ? v : 0.f;
              Cu[idx] = f2bf(rv);
              lsum += rv * rv;
            } else { // MODE 4: normal + per-batch transposed store
              unsigned short pv = f2bf(v);
              Cu[idx] = pv;
              unsigned bb = (unsigned)m / (unsigned)N_;
              unsigned mm = (unsigned)m - bb * N_;
              C2[((size_t)bb * D_ + n) * TLD + mm] = pv;
            }
          }
        }
      }
    }
  }
  if (MODE == 2 || MODE == 3) {
    if (lsum != 0.f) atomicAdd(e_slot, -0.5f * lsum);
  }
}

// ---------------------------------------------------------------------------
// Block reduction helpers (256 threads)
// ---------------------------------------------------------------------------
__device__ __forceinline__ float bsum(float v, float* red) {
  int tid = threadIdx.x;
  red[tid] = v; __syncthreads();
  for (int o = 128; o > 0; o >>= 1) { if (tid < o) red[tid] += red[tid + o]; __syncthreads(); }
  float r = red[0]; __syncthreads();
  return r;
}
__device__ __forceinline__ float bmax(float v, float* red) {
  int tid = threadIdx.x;
  red[tid] = v; __syncthreads();
  for (int o = 128; o > 0; o >>= 1) { if (tid < o) red[tid] = fmaxf(red[tid], red[tid + o]); __syncthreads(); }
  float r = red[0]; __syncthreads();
  return r;
}

// ---------------------------------------------------------------------------
// LayerNorm forward: g = gamma*(x-mu)*rstd + delta (bf16 out), save mu/rstd
// ---------------------------------------------------------------------------
__global__ __launch_bounds__(256)
void ln_fwd(const float* __restrict__ x, const float* __restrict__ gamma,
            const float* __restrict__ delta, unsigned short* __restrict__ g,
            float* __restrict__ meanv, float* __restrict__ rstdv)
{
  const int t = blockIdx.x, tid = threadIdx.x;
  __shared__ float red[256];
  const float* xp = x + (size_t)t * D_;
  float v[3];
  #pragma unroll
  for (int i = 0; i < 3; ++i) v[i] = xp[tid + i * 256];
  float mu = bsum(v[0] + v[1] + v[2], red) * (1.f / D_);
  float s2 = 0.f;
  #pragma unroll
  for (int i = 0; i < 3; ++i) { float d = v[i] - mu; s2 += d * d; }
  float var = bsum(s2, red) * (1.f / D_);
  float rs = rsqrtf(var + LN_EPS_C);
  unsigned short* gp = g + (size_t)t * D_;
  #pragma unroll
  for (int i = 0; i < 3; ++i) {
    int c = tid + i * 256;
    gp[c] = f2bf(gamma[c] * (v[i] - mu) * rs + delta[c]);
  }
  if (tid == 0) { meanv[t] = mu; rstdv[t] = rs; }
}

// LayerNorm backward: dx = rs*(a - mean(a) - xhat*mean(a*xhat)), a = dg*gamma
__global__ __launch_bounds__(256)
void ln_bwd(const float* __restrict__ x, const float* __restrict__ dG,
            const float* __restrict__ gamma, const float* __restrict__ meanv,
            const float* __restrict__ rstdv, float* __restrict__ dx)
{
  const int t = blockIdx.x, tid = threadIdx.x;
  __shared__ float red[256];
  const float* xp = x + (size_t)t * D_;
  const float* dp = dG + (size_t)t * D_;
  float mu = meanv[t], rs = rstdv[t];
  float a[3], xh[3], s1 = 0.f, s2 = 0.f;
  #pragma unroll
  for (int i = 0; i < 3; ++i) {
    int c = tid + i * 256;
    xh[i] = (xp[c] - mu) * rs;
    a[i]  = dp[c] * gamma[c];
    s1 += a[i]; s2 += a[i] * xh[i];
  }
  s1 = bsum(s1, red) * (1.f / D_);
  s2 = bsum(s2, red) * (1.f / D_);
  #pragma unroll
  for (int i = 0; i < 3; ++i) {
    int c = tid + i * 256;
    dx[(size_t)t * D_ + c] = rs * (a[i] - s1 - xh[i] * s2);
  }
}

// ---------------------------------------------------------------------------
// Row softmax / logsumexp over S[row, 0..195] (row = (b*H+h)*196 + q).
// e_slot += -(1/beta)*lse; optionally write P (ld TLD) and P^T (per-bh, ld TLD)
// ---------------------------------------------------------------------------
template<bool WRITE_P>
__global__ __launch_bounds__(256)
void attn_rows(const float* __restrict__ S, unsigned short* __restrict__ P,
               unsigned short* __restrict__ Pt, float* __restrict__ e_slot)
{
  const int row = blockIdx.x, tid = threadIdx.x;
  __shared__ float red[256];
  const float* sp = S + (size_t)row * N_;
  float v  = (tid < N_) ? sp[tid] : -3.4e38f;
  float mx = bmax(v, red);
  float e  = (tid < N_) ? __expf(v - mx) : 0.f;
  float sum = bsum(e, red);
  if (WRITE_P && tid < N_) {
    unsigned short pv = f2bf(e / sum);
    int bh = row / N_, q = row % N_;
    P[(size_t)row * TLD + tid] = pv;
    Pt[((size_t)bh * N_ + tid) * TLD + q] = pv;
  }
  if (tid == 0) atomicAdd(e_slot, -(1.f / BETA_C) * (mx + __logf(sum)));
}

// ---------------------------------------------------------------------------
// Tiled transpose: dst[c*ldd + r] = bf16(src[r*cols + c])   (f32 -> bf16)
// ---------------------------------------------------------------------------
__global__ __launch_bounds__(256)
void k_transpose_bf(unsigned short* __restrict__ dst, const float* __restrict__ src,
                    int rows, int cols, int ldd)
{
  __shared__ float tile[32][33];
  const int c0 = blockIdx.x * 32, r0 = blockIdx.y * 32;
  const int tx = threadIdx.x, ty = threadIdx.y;       // 32 x 8
  #pragma unroll
  for (int i = 0; i < 32; i += 8) {
    int r = r0 + ty + i, c = c0 + tx;
    tile[ty + i][tx] = (r < rows && c < cols) ? src[(size_t)r * cols + c] : 0.f;
  }
  __syncthreads();
  #pragma unroll
  for (int i = 0; i < 32; i += 8) {
    int r = c0 + ty + i, c = r0 + tx;                 // transposed coords
    if (r < cols && c < rows) dst[(size_t)r * ldd + c] = f2bf(tile[tx][ty + i]);
  }
}

// ---------------------------------------------------------------------------
// Small utility / control kernels
// scal: [0]=lr0 [1]=ss [2]=sy [3]=chosen [4]=e_ref [5..8]=ne[0..3]
// ---------------------------------------------------------------------------
__global__ void k_zero(float* p, int n) {
  int i = blockIdx.x * blockDim.x + threadIdx.x;
  if (i < n) p[i] = 0.f;
}
__global__ void k_copy(float* dst, const float* src, size_t n) {
  size_t i = (size_t)blockIdx.x * blockDim.x + threadIdx.x;
  if (i < n) dst[i] = src[i];
}
__global__ void k_f2bf(unsigned short* dst, const float* src, size_t n) {
  size_t i = (size_t)blockIdx.x * blockDim.x + threadIdx.x;
  if (i < n) dst[i] = f2bf(src[i]);
}
__global__ void bb_dots(const float* x, const float* px, const float* g,
                        const float* pg, float* scal, size_t n) {
  __shared__ float r1[256], r2[256];
  float ss = 0.f, sy = 0.f;
  for (size_t i = (size_t)blockIdx.x * blockDim.x + threadIdx.x; i < n;
       i += (size_t)gridDim.x * blockDim.x) {
    float s = x[i] - px[i], y = g[i] - pg[i];
    ss += s * s; sy += s * y;
  }
  int tid = threadIdx.x;
  r1[tid] = ss; r2[tid] = sy; __syncthreads();
  for (int o = 128; o > 0; o >>= 1) {
    if (tid < o) { r1[tid] += r1[tid + o]; r2[tid] += r2[tid + o]; }
    __syncthreads();
  }
  if (tid == 0) { atomicAdd(&scal[1], r1[0]); atomicAdd(&scal[2], r2[0]); }
}
__global__ void k_lr0(float* scal, int t) {
  scal[0] = (t == 0) ? 1.0f : scal[1] / fmaxf(scal[2], 1e-8f);
}
__global__ void k_choose(float* scal) {
  float lr0 = scal[0], eref = scal[4];
  float chosen = lr0 * 0.0625f;                 // gamma^4 fallback
  const float pw[4] = {1.f, 0.5f, 0.25f, 0.125f};
  for (int k = 3; k >= 0; --k)                  // last write wins => smallest k
    if (scal[5 + k] < eref) chosen = lr0 * pw[k];
  scal[3] = chosen;
}
__global__ void k_trial(float* xt, const float* x, const float* g,
                        const float* scal, float ck, size_t n) {
  size_t i = (size_t)blockIdx.x * blockDim.x + threadIdx.x;
  if (i < n) xt[i] = x[i] - (scal[0] * ck) * g[i];
}
__global__ void k_update(float* x, const float* g, float* px, float* pg,
                         const float* scal, size_t n) {
  size_t i = (size_t)blockIdx.x * blockDim.x + threadIdx.x;
  if (i < n) {
    float xv = x[i], gv = g[i];
    px[i] = xv; pg[i] = gv;
    x[i] = xv - scal[3] * gv;
  }
}

// ---------------------------------------------------------------------------
extern "C" void kernel_launch(void* const* d_in, const int* in_sizes, int n_in,
                              void* d_out, int out_size, void* d_ws, size_t ws_size,
                              hipStream_t stream)
{
  (void)in_sizes; (void)n_in; (void)out_size; (void)ws_size;
  const float* x_in  = (const float*)d_in[0];
  const float* gamma = (const float*)d_in[1];
  const float* delta = (const float*)d_in[2];
  const float* wk    = (const float*)d_in[3];
  const float* wq    = (const float*)d_in[4];
  const float* xi    = (const float*)d_in[5];

  char* ws = (char*)d_ws; size_t off = 0;
  auto alloc = [&](size_t bytes) -> char* {
    char* p = ws + off; off = (off + bytes + 255) & ~(size_t)255; return p;
  };
  float* x_cur   = (float*)alloc(TD * 4);
  float* gradb   = (float*)alloc(TD * 4);
  float* prev_x  = (float*)alloc(TD * 4);
  float* prev_g  = (float*)alloc(TD * 4);
  float* x_trial = (float*)alloc(TD * 4);
  float* dGbuf   = (float*)alloc(TD * 4);
  float* Sbuf    = (float*)alloc((size_t)B_ * H_ * N_ * N_ * 4);
  float* meanv   = (float*)alloc(T_ * 4);
  float* rstdv   = (float*)alloc(T_ * 4);
  float* scal    = (float*)alloc(16 * 4);
  unsigned short* g_bf    = (unsigned short*)alloc(TD * 2);
  unsigned short* wk_bf   = (unsigned short*)alloc((size_t)D_ * D_ * 2);
  unsigned short* wq_bf   = (unsigned short*)alloc((size_t)D_ * D_ * 2);
  unsigned short* xi_bf   = (unsigned short*)alloc((size_t)M_ * D_ * 2);
  unsigned short* wkT_bf  = (unsigned short*)alloc((size_t)D_ * D_ * 2);
  unsigned short* wqT_bf  = (unsigned short*)alloc((size_t)D_ * D_ * 2);
  unsigned short* xiT_bf  = (unsigned short*)alloc((size_t)D_ * M_ * 2);
  unsigned short* Kall_bf = (unsigned short*)alloc(TD * 2);
  unsigned short* Qall_bf = (unsigned short*)alloc(TD * 2);
  unsigned short* KallT   = (unsigned short*)alloc((size_t)B_ * D_ * TLD * 2);
  unsigned short* QallT   = (unsigned short*)alloc((size_t)B_ * D_ * TLD * 2);
  unsigned short* dQ_bf   = (unsigned short*)alloc(TD * 2);
  unsigned short* dK_bf   = (unsigned short*)alloc(TD * 2);
  unsigned short* P_bf    = (unsigned short*)alloc((size_t)B_ * H_ * N_ * TLD * 2);
  unsigned short* Pt_bf   = (unsigned short*)alloc((size_t)B_ * H_ * N_ * TLD * 2);
  unsigned short* h_bf    = (unsigned short*)alloc((size_t)T_ * M_ * 2);

  const int EW = (int)((TD + 255) / 256);
  const long STRD = (long)N_ * D_;             // token stride per batch in K/Q/dQ/dK
  const long SPB  = (long)N_ * TLD;            // P / Pt per-(b,h) stride
  const long STB  = (long)D_ * TLD;            // KallT/QallT per-batch stride

  // Energy (and optionally gradient) pass at xin; energy -> *e_slot
  auto energy_pass = [&](const float* xin, float* e_slot, bool full) {
    k_zero<<<1, 1, 0, stream>>>(e_slot, 1);
    ln_fwd<<<T_, 256, 0, stream>>>(xin, gamma, delta, g_bf, meanv, rstdv);
    // K = G @ Wk^T (plus per-batch K^T), same for Q   (NT, MODE 4)
    gemm_nt<false, 4><<<dim3(25, 12, 1), 128, 0, stream>>>(
        (const __bf16*)g_bf, (const __bf16*)wk_bf, Kall_bf, KallT,
        T_, D_, D_, D_, D_, D_, 0, 0, 0, 0, 0, 0, 1, 1.0f, nullptr);
    gemm_nt<false, 4><<<dim3(25, 12, 1), 128, 0, stream>>>(
        (const __bf16*)g_bf, (const __bf16*)wq_bf, Qall_bf, QallT,
        T_, D_, D_, D_, D_, D_, 0, 0, 0, 0, 0, 0, 1, 1.0f, nullptr);
    // S[b,h] = beta * Q_bh @ K_bh^T  (NT batched, f32)
    gemm_nt<false, 0><<<dim3(4, 4, B_ * H_), 128, 0, stream>>>(
        (const __bf16*)Qall_bf, (const __bf16*)Kall_bf, Sbuf, nullptr,
        N_, N_, Y_, D_, D_, N_,
        STRD, Y_, STRD, Y_, (long)H_ * N_ * N_, (long)N_ * N_, H_, BETA_C, nullptr);
    // logsumexp rows (+softmax P, P^T when full)
    if (full) attn_rows<true><<<B_ * H_ * N_, 256, 0, stream>>>(Sbuf, P_bf, Pt_bf, e_slot);
    else      attn_rows<false><<<B_ * H_ * N_, 256, 0, stream>>>(Sbuf, nullptr, nullptr, e_slot);
    // Hopfield: relu(G @ Xi^T); energy fused in epilogue, store h only when full
    if (full)
      gemm_nt<false, 3><<<dim3(25, 48, 1), 128, 0, stream>>>(
          (const __bf16*)g_bf, (const __bf16*)xi_bf, h_bf, nullptr,
          T_, M_, D_, D_, D_, M_, 0, 0, 0, 0, 0, 0, 1, 1.0f, e_slot);
    else
      gemm_nt<false, 2><<<dim3(25, 48, 1), 128, 0, stream>>>(
          (const __bf16*)g_bf, (const __bf16*)xi_bf, nullptr, nullptr,
          T_, M_, D_, D_, D_, M_, 0, 0, 0, 0, 0, 0, 1, 1.0f, e_slot);

    if (full) {
      // dQ_bh = -P_bh @ (K_bh^T)^T  : NT with A=P (ld TLD), B=KallT slice (ld TLD)
      gemm_nt<false, 1><<<dim3(4, 1, B_ * H_), 128, 0, stream>>>(
          (const __bf16*)P_bf, (const __bf16*)KallT, dQ_bf, nullptr,
          N_, Y_, N_, TLD, TLD, D_,
          (long)H_ * SPB, SPB, STB, (long)Y_ * TLD, STRD, Y_, H_, -1.0f, nullptr);
      // dK_bh = -P_bh^T @ Q_bh : NT with A=Pt, B=QallT slice
      gemm_nt<false, 1><<<dim3(4, 1, B_ * H_), 128, 0, stream>>>(
          (const __bf16*)Pt_bf, (const __bf16*)QallT, dK_bf, nullptr,
          N_, Y_, N_, TLD, TLD, D_,
          (long)H_ * SPB, SPB, STB, (long)Y_ * TLD, STRD, Y_, H_, -1.0f, nullptr);
      // dG = dK @ Wk + dQ @ Wq - h @ Xi   (all NT via pre-transposed weights)
      gemm_nt<false, 0><<<dim3(25, 12, 1), 128, 0, stream>>>(
          (const __bf16*)dK_bf, (const __bf16*)wkT_bf, dGbuf, nullptr,
          T_, D_, D_, D_, D_, D_, 0, 0, 0, 0, 0, 0, 1, 1.0f, nullptr);
      gemm_nt<true, 0><<<dim3(25, 12, 1), 128, 0, stream>>>(
          (const __bf16*)dQ_bf, (const __bf16*)wqT_bf, dGbuf, nullptr,
          T_, D_, D_, D_, D_, D_, 0, 0, 0, 0, 0, 0, 1, 1.0f, nullptr);
      gemm_nt<true, 0><<<dim3(25, 12, 1), 128, 0, stream>>>(
          (const __bf16*)h_bf, (const __bf16*)xiT_bf, dGbuf, nullptr,
          T_, D_, M_, M_, M_, D_, 0, 0, 0, 0, 0, 0, 1, -1.0f, nullptr);
      ln_bwd<<<T_, 256, 0, stream>>>(xin, dGbuf, gamma, meanv, rstdv, gradb);
    }
  };

  // ---- init: copy x, bf16 weights + transposed weights ----
  k_copy<<<EW, 256, 0, stream>>>(x_cur, x_in, TD);
  k_f2bf<<<(D_ * D_ + 255) / 256, 256, 0, stream>>>(wk_bf, wk, (size_t)D_ * D_);
  k_f2bf<<<(D_ * D_ + 255) / 256, 256, 0, stream>>>(wq_bf, wq, (size_t)D_ * D_);
  k_f2bf<<<(int)(((size_t)M_ * D_ + 255) / 256), 256, 0, stream>>>(xi_bf, xi, (size_t)M_ * D_);
  k_transpose_bf<<<dim3(24, 24), dim3(32, 8), 0, stream>>>(wkT_bf, wk, D_, D_, D_);
  k_transpose_bf<<<dim3(24, 24), dim3(32, 8), 0, stream>>>(wqT_bf, wq, D_, D_, D_);
  k_transpose_bf<<<dim3(24, 96), dim3(32, 8), 0, stream>>>(xiT_bf, xi, M_, D_, M_);

  // ---- 12 descent steps: value+grad, BB lr0, 4 Armijo trials, select, update ----
  const float pw[4] = {1.f, 0.5f, 0.25f, 0.125f};
  for (int t = 0; t < 12; ++t) {
    energy_pass(x_cur, &scal[4], true);          // e_ref + grad
    if (t > 0) {
      k_zero<<<1, 1, 0, stream>>>(scal + 1, 2);
      bb_dots<<<512, 256, 0, stream>>>(x_cur, prev_x, gradb, prev_g, scal, TD);
    }
    k_lr0<<<1, 1, 0, stream>>>(scal, t);
    for (int k = 0; k < 4; ++k) {
      k_trial<<<EW, 256, 0, stream>>>(x_trial, x_cur, gradb, scal, pw[k], TD);
      energy_pass(x_trial, &scal[5 + k], false);
    }
    k_choose<<<1, 1, 0, stream>>>(scal);
    k_update<<<EW, 256, 0, stream>>>(x_cur, gradb, prev_x, prev_g, scal, TD);
  }
  k_copy<<<EW, 256, 0, stream>>>((float*)d_out, x_cur, TD);
}
